// AgentAttention_50483045597628
// MI455X (gfx1250) — matrix operand order repaired and checked by reading, not compiled, not run on hardware
//
#include <hip/hip_runtime.h>
#include <math.h>

// Problem constants (from reference)
#define B_   16
#define N_   3136
#define C_   256
#define H_   8
#define A_   49
#define HW_  56
#define DH_  32
#define SCALE_ 0.17677669529663687f   // 32^-0.5

typedef __attribute__((ext_vector_type(16))) __bf16 v16bf;
typedef __attribute__((ext_vector_type(8)))  __bf16 v8bf;
typedef __attribute__((ext_vector_type(8)))  float  v8f;

__device__ __forceinline__ __bf16 f2bf(float f) {
  unsigned u = __builtin_bit_cast(unsigned, f);
  u += 0x7FFFu + ((u >> 16) & 1u);          // round-to-nearest-even
  unsigned short h = (unsigned short)(u >> 16);
  return __builtin_bit_cast(__bf16, h);
}
__device__ __forceinline__ float bf2f(__bf16 b) {
  unsigned short h = __builtin_bit_cast(unsigned short, b);
  unsigned u = ((unsigned)h) << 16;
  return __builtin_bit_cast(float, u);
}

// ---------- elementwise f32 -> bf16 ----------
__global__ void k_f2b(const float* __restrict__ s, __bf16* __restrict__ d, size_t n) {
  size_t i = (size_t)blockIdx.x * blockDim.x + threadIdx.x;
  if (i < n) d[i] = f2bf(s[i]);
}

// ---------- bf16 WMMA GEMM, 2x2 register-blocked ----------
// out[m][n] = sum_k A[m][k]*W[n][k] (+bias[n])
// A: M x K bf16 row-major. W: N x K bf16 row-major (== column-major B-matrix).
// grid = (M/32, N/256), block = 256 (8 waves); wave computes a 32x32 output block
// (4 accumulators), so 4 operand loads feed 4 WMMAs per K-step (2x intensity).
// Shapes per CDNA5 ISA 7.12.2 (16-bit A 16x32, B 32x16, f32 C/D 16x16); A/B k order is contiguous, not ISA's.
__global__ void k_gemm(const __bf16* __restrict__ A, const __bf16* __restrict__ W,
                       const float* __restrict__ bias,
                       float* __restrict__ outf, __bf16* __restrict__ outb,
                       int N, int K) {
  const int lane = threadIdx.x & 31;
  const int wv   = threadIdx.x >> 5;
  const int l16  = lane & 15;
  const int hl   = lane >> 4;                 // 0 = lanes 0-15, 1 = lanes 16-31
  const int m0 = blockIdx.x * 32;
  const int n0 = blockIdx.y * 256 + wv * 32;

  const __bf16* arow0 = A + (size_t)(m0 + l16) * K;
  const __bf16* arow1 = A + (size_t)(m0 + 16 + l16) * K;
  const __bf16* wrow0 = W + (size_t)(n0 + l16) * K + hl * 16;
  const __bf16* wrow1 = W + (size_t)(n0 + 16 + l16) * K + hl * 16;

  v8f acc00 = {}, acc01 = {}, acc10 = {}, acc11 = {};
  for (int k0 = 0; k0 < K; k0 += 32) {
    // A lane fragments: contiguous K = k0 + hl*16 + {0..15}
    v8bf a0lo = *(const v8bf*)(arow0 + k0 + hl * 16);
    v8bf a0hi = *(const v8bf*)(arow0 + k0 + 8 + hl * 16);
    v8bf a1lo = *(const v8bf*)(arow1 + k0 + hl * 16);
    v8bf a1hi = *(const v8bf*)(arow1 + k0 + 8 + hl * 16);
    v16bf av0 = __builtin_shufflevector(a0lo, a0hi, 0,1,2,3,4,5,6,7,8,9,10,11,12,13,14,15);
    v16bf av1 = __builtin_shufflevector(a1lo, a1hi, 0,1,2,3,4,5,6,7,8,9,10,11,12,13,14,15);
    // B lane fragments: contiguous W[n][k0 + hl*16 .. +15]
    v16bf bv0 = *(const v16bf*)(wrow0 + k0);
    v16bf bv1 = *(const v16bf*)(wrow1 + k0);
    acc00 = __builtin_amdgcn_wmma_f32_16x16x32_bf16(false, av0, false, bv0, (short)0, acc00, false, false);
    acc01 = __builtin_amdgcn_wmma_f32_16x16x32_bf16(false, av0, false, bv1, (short)0, acc01, false, false);
    acc10 = __builtin_amdgcn_wmma_f32_16x16x32_bf16(false, av1, false, bv0, (short)0, acc10, false, false);
    acc11 = __builtin_amdgcn_wmma_f32_16x16x32_bf16(false, av1, false, bv1, (short)0, acc11, false, false);
  }
  // D layout: m = tile_m + r + (lane>=16 ? 8 : 0), n = tile_n + (lane&15)
  const int nA = n0 + l16, nB = n0 + 16 + l16;
  const float bvA = bias ? bias[nA] : 0.0f;
  const float bvB = bias ? bias[nB] : 0.0f;
#pragma unroll
  for (int r = 0; r < 8; ++r) {
    int mA = m0 + r + hl * 8;
    int mB = m0 + 16 + r + hl * 8;
    float v00 = acc00[r] + bvA, v01 = acc01[r] + bvB;
    float v10 = acc10[r] + bvA, v11 = acc11[r] + bvB;
    if (outf) {
      outf[(size_t)mA * N + nA] = v00;
      outf[(size_t)mA * N + nB] = v01;
      outf[(size_t)mB * N + nA] = v10;
      outf[(size_t)mB * N + nB] = v11;
    } else {
      outb[(size_t)mA * N + nA] = f2bf(v00);
      outb[(size_t)mA * N + nB] = f2bf(v01);
      outb[(size_t)mB * N + nA] = f2bf(v10);
      outb[(size_t)mB * N + nB] = f2bf(v11);
    }
  }
}

// ---------- adaptive avg-pool 56x56 -> 7x7, reshaped to heads ----------
// agents[((b*8+h)*49+a)*32+d] = mean over 8x8 window of q(b, n, c=h*32+d)
__global__ void k_pool(const __bf16* __restrict__ q, float* __restrict__ ag) {
  int idx = blockIdx.x * blockDim.x + threadIdx.x;
  if (idx >= B_ * A_ * C_) return;
  int c = idx % C_;
  int a = (idx / C_) % A_;
  int b = idx / (C_ * A_);
  int py = a / 7, px = a % 7;
  float s = 0.f;
  for (int dy = 0; dy < 8; ++dy)
    for (int dx = 0; dx < 8; ++dx) {
      int n = (py * 8 + dy) * HW_ + px * 8 + dx;
      s += bf2f(q[((size_t)b * N_ + n) * C_ + c]);
    }
  int h = c / DH_, d = c % DH_;
  ag[((size_t)(b * H_ + h) * A_ + a) * DH_ + d] = s * (1.0f / 64.0f);
}

// ---------- bilinear 7x7 -> 56x56 sample (half-pixel, edge clamp) ----------
__device__ __forceinline__ float bilin7(const float* __restrict__ t, int y, int x) {
  float sy = (y + 0.5f) * 0.125f - 0.5f;
  float sx = (x + 0.5f) * 0.125f - 0.5f;
  float fy0 = floorf(sy), fx0 = floorf(sx);
  float fy = sy - fy0, fx = sx - fx0;
  int y0 = (int)fy0, x0 = (int)fx0;
  int y1 = y0 + 1, x1 = x0 + 1;
  y0 = y0 < 0 ? 0 : (y0 > 6 ? 6 : y0);
  y1 = y1 < 0 ? 0 : (y1 > 6 ? 6 : y1);
  x0 = x0 < 0 ? 0 : (x0 > 6 ? 6 : x0);
  x1 = x1 < 0 ? 0 : (x1 > 6 ? 6 : x1);
  return (1.f - fy) * ((1.f - fx) * t[y0 * 7 + x0] + fx * t[y0 * 7 + x1]) +
         fy         * ((1.f - fx) * t[y1 * 7 + x0] + fx * t[y1 * 7 + x1]);
}

// bias1[h][a][n] = resize(an_bias)(h,a,y,x) + ah(h,a,y) + aw(h,a,x)
__global__ void k_bias1(const float* __restrict__ an, const float* __restrict__ ah,
                        const float* __restrict__ aw, float* __restrict__ b1) {
  int idx = blockIdx.x * blockDim.x + threadIdx.x;
  if (idx >= H_ * A_ * N_) return;
  int n = idx % N_;
  int a = (idx / N_) % A_;
  int h = idx / (N_ * A_);
  int y = n / HW_, x = n % HW_;
  float v = bilin7(an + (size_t)(h * A_ + a) * 49, y, x);
  v += ah[(size_t)(h * A_ + a) * HW_ + y];
  v += aw[(size_t)(h * A_ + a) * HW_ + x];
  b1[idx] = v;
}

// bias2[h][n][a] = resize(na_bias)(h,a,y,x) + ha(h,y,a) + wa(h,x,a)
__global__ void k_bias2(const float* __restrict__ na, const float* __restrict__ ha,
                        const float* __restrict__ wa, float* __restrict__ b2) {
  int idx = blockIdx.x * blockDim.x + threadIdx.x;
  if (idx >= H_ * N_ * A_) return;
  int a = idx % A_;
  int n = (idx / A_) % N_;
  int h = idx / (A_ * N_);
  int y = n / HW_, x = n % HW_;
  float v = bilin7(na + (size_t)(h * A_ + a) * 49, y, x);
  v += ha[((size_t)h * HW_ + y) * A_ + a];
  v += wa[((size_t)h * HW_ + x) * A_ + a];
  b2[idx] = v;
}

// ---------- stage 1: agent tokens attend to k/v, online softmax over N keys ----------
// one wave per (b,h,a); LANE-PER-KEY: lane l handles keys l, l+32, ... with
// aligned 16B v8bf loads of the k/v rows; per-lane online softmax, single
// cross-lane butterfly merge at the end. kv layout (b,n,2C): k at c, v at C+c.
__global__ void k_stage1(const float* __restrict__ agents, const __bf16* __restrict__ kv,
                         const float* __restrict__ b1, float* __restrict__ agv) {
  int wid = (blockIdx.x * blockDim.x + threadIdx.x) >> 5;
  if (wid >= B_ * H_ * A_) return;
  int lane = threadIdx.x & 31;
  int a = wid % A_;
  int bh = wid / A_;
  int h = bh % H_, b = bh / H_;

  // agent vector (wave-uniform address), pre-scaled
  float av[DH_];
  const float* ap = agents + ((size_t)(b * H_ + h) * A_ + a) * DH_;
#pragma unroll
  for (int d = 0; d < DH_; ++d) av[d] = ap[d] * SCALE_;

  const __bf16* kvb = kv + (size_t)b * N_ * (2 * C_) + h * DH_;
  const float* bb = b1 + (size_t)(h * A_ + a) * N_;

  float m = -INFINITY, s = 0.f;
  float acc[DH_];
#pragma unroll
  for (int d = 0; d < DH_; ++d) acc[d] = 0.f;

  for (int key = lane; key < N_; key += 32) {
    const __bf16* kp = kvb + (size_t)key * (2 * C_);
    const v8bf* k8 = (const v8bf*)kp;
    float dot = bb[key];
#pragma unroll
    for (int i = 0; i < 4; ++i) {
      v8bf kk = k8[i];
#pragma unroll
      for (int j = 0; j < 8; ++j) dot += av[i * 8 + j] * bf2f(kk[j]);
    }
    float mn = fmaxf(m, dot);
    float corr = __expf(m - mn);
    float e = __expf(dot - mn);
    s = s * corr + e;
    const v8bf* v8 = (const v8bf*)(kp + C_);
#pragma unroll
    for (int i = 0; i < 4; ++i) {
      v8bf vv = v8[i];
#pragma unroll
      for (int j = 0; j < 8; ++j)
        acc[i * 8 + j] = acc[i * 8 + j] * corr + e * bf2f(vv[j]);
    }
    m = mn;
  }

  // cross-lane online-softmax merge (butterfly)
#pragma unroll
  for (int off = 16; off >= 1; off >>= 1) {
    float m2 = __shfl_xor(m, off, 32);
    float s2 = __shfl_xor(s, off, 32);
    float mn = fmaxf(m, m2);
    float c1 = __expf(m - mn), c2 = __expf(m2 - mn);
    s = s * c1 + s2 * c2;
#pragma unroll
    for (int d = 0; d < DH_; ++d) {
      float o = __shfl_xor(acc[d], off, 32);
      acc[d] = acc[d] * c1 + o * c2;
    }
    m = mn;
  }
  if (lane == 0) {
    float inv = 1.0f / s;
    float* op = agv + ((size_t)(b * H_ + h) * A_ + a) * DH_;
#pragma unroll
    for (int d = 0; d < DH_; ++d) op[d] = acc[d] * inv;
  }
}

// ---------- stage 2: queries attend to 49 agents (softmax over a) ----------
// grid = (ceil(N/256), H, B); one thread per query row per head.
__global__ void k_stage2(const __bf16* __restrict__ q, const float* __restrict__ agents,
                         const float* __restrict__ agv, const float* __restrict__ b2,
                         float* __restrict__ out) {
  __shared__ float sA[A_ * DH_];
  __shared__ float sV[A_ * DH_];
  int b = blockIdx.z, h = blockIdx.y;
  int tid = threadIdx.x;
  const size_t agbase = (size_t)(b * H_ + h) * A_ * DH_;
  for (int i = tid; i < A_ * DH_; i += blockDim.x) {
    sA[i] = agents[agbase + i];
    sV[i] = agv[agbase + i];
  }
  __syncthreads();
  int n = blockIdx.x * blockDim.x + tid;
  if (n >= N_) return;

  float qd[DH_];
  const v8bf* qp = (const v8bf*)(q + ((size_t)b * N_ + n) * C_ + h * DH_);
#pragma unroll
  for (int i = 0; i < 4; ++i) {
    v8bf t = qp[i];
#pragma unroll
    for (int j = 0; j < 8; ++j) qd[i * 8 + j] = bf2f(t[j]) * SCALE_;
  }
  const float* bb = b2 + ((size_t)h * N_ + n) * A_;

  float mx = -INFINITY;
  for (int a = 0; a < A_; ++a) {
    float l = bb[a];
#pragma unroll
    for (int d = 0; d < DH_; ++d) l += qd[d] * sA[a * DH_ + d];
    mx = fmaxf(mx, l);
  }
  float s = 0.f;
  float o[DH_];
#pragma unroll
  for (int d = 0; d < DH_; ++d) o[d] = 0.f;
  for (int a = 0; a < A_; ++a) {
    float l = bb[a];
#pragma unroll
    for (int d = 0; d < DH_; ++d) l += qd[d] * sA[a * DH_ + d];
    float p = __expf(l - mx);
    s += p;
#pragma unroll
    for (int d = 0; d < DH_; ++d) o[d] += p * sV[a * DH_ + d];
  }
  float inv = 1.0f / s;
  float* op = out + ((size_t)b * N_ + n) * C_ + h * DH_;
#pragma unroll
  for (int d = 0; d < DH_; ++d) op[d] = o[d] * inv;
}

// ---------- 3x3 depthwise conv on v (from kv), added into branch output ----------
__global__ void k_dwc(const __bf16* __restrict__ kv, const float* __restrict__ w,
                      const float* __restrict__ bias, float* __restrict__ out) {
  size_t idx = (size_t)blockIdx.x * blockDim.x + threadIdx.x;
  if (idx >= (size_t)B_ * N_ * C_) return;
  int c = (int)(idx % C_);
  int n = (int)((idx / C_) % N_);
  int b = (int)(idx / ((size_t)C_ * N_));
  int y = n / HW_, x = n % HW_;
  float sum = bias[c];
  for (int ky = 0; ky < 3; ++ky) {
    int yy = y + ky - 1;
    if (yy < 0 || yy >= HW_) continue;
    for (int kx = 0; kx < 3; ++kx) {
      int xx = x + kx - 1;
      if (xx < 0 || xx >= HW_) continue;
      float vv = bf2f(kv[((size_t)b * N_ + yy * HW_ + xx) * (2 * C_) + C_ + c]);
      sum += w[(ky * 3 + kx) * C_ + c] * vv;
    }
  }
  out[idx] += sum;
}

// ===========================================================================
extern "C" void kernel_launch(void* const* d_in, const int* in_sizes, int n_in,
                              void* d_out, int out_size, void* d_ws, size_t ws_size,
                              hipStream_t stream) {
  (void)in_sizes; (void)n_in; (void)out_size; (void)ws_size;
  const float* x = (const float*)d_in[0];
  const float* y = (const float*)d_in[1];
  // params: 0 q_w, 1 kv_w, 2 proj_w, 3 proj_b, 4 dwc_w, 5 dwc_b,
  //         6 an, 7 na, 8 ah, 9 aw, 10 ha, 11 wa   (per modality)
  const float* P[2][12];
  for (int m = 0; m < 2; ++m)
    for (int j = 0; j < 12; ++j)
      P[m][j] = (const float*)d_in[4 + m * 12 + j];
  float* out = (float*)d_out;

  // ---- workspace layout (~219 MB total) ----
  char* ws = (char*)d_ws;
  size_t off = 0;
  auto take = [&](size_t bytes) { size_t o = off; off = (off + bytes + 255) & ~(size_t)255; return o; };
  const size_t MN  = (size_t)B_ * N_;
  const size_t MNC = MN * C_;
  __bf16* xb   = (__bf16*)(ws + take(MNC * 2));
  __bf16* yb   = (__bf16*)(ws + take(MNC * 2));
  __bf16* qx   = (__bf16*)(ws + take(MNC * 2));
  __bf16* qy   = (__bf16*)(ws + take(MNC * 2));
  __bf16* kvb  = (__bf16*)(ws + take(MNC * 2 * 2));       // (B,N,2C)
  float*  ob   = (float*) (ws + take(MNC * 4));
  float*  b1   = (float*) (ws + take((size_t)H_ * A_ * N_ * 4));
  float*  b2   = (float*) (ws + take((size_t)H_ * N_ * A_ * 4));
  float*  ag0  = (float*) (ws + take((size_t)B_ * H_ * A_ * DH_ * 4));
  float*  ag1  = (float*) (ws + take((size_t)B_ * H_ * A_ * DH_ * 4));
  float*  agv  = (float*) (ws + take((size_t)B_ * H_ * A_ * DH_ * 4));
  __bf16* wq   = (__bf16*)(ws + take((size_t)C_ * C_ * 2));
  __bf16* wkv  = (__bf16*)(ws + take((size_t)2 * C_ * C_ * 2));
  __bf16* wpr  = (__bf16*)(ws + take((size_t)C_ * C_ * 2));

  auto cvt = [&](const float* s, __bf16* d, size_t n) {
    k_f2b<<<dim3((unsigned)((n + 255) / 256)), dim3(256), 0, stream>>>(s, d, n);
  };

  // inputs -> bf16
  cvt(x, xb, MNC);
  cvt(y, yb, MNC);

  // q projections (WMMA) + agent pooling for both modalities
  cvt(P[0][0], wq, (size_t)C_ * C_);
  k_gemm<<<dim3(MN / 32, C_ / 256), dim3(256), 0, stream>>>(xb, wq, nullptr, nullptr, qx, C_, C_);
  k_pool<<<dim3((B_ * A_ * C_ + 255) / 256), dim3(256), 0, stream>>>(qx, ag0);
  cvt(P[1][0], wq, (size_t)C_ * C_);
  k_gemm<<<dim3(MN / 32, C_ / 256), dim3(256), 0, stream>>>(yb, wq, nullptr, nullptr, qy, C_, C_);
  k_pool<<<dim3((B_ * A_ * C_ + 255) / 256), dim3(256), 0, stream>>>(qy, ag1);

  for (int m = 0; m < 2; ++m) {
    __bf16* src     = m ? yb : xb;
    __bf16* qq      = m ? qy : qx;
    float*  agOther = m ? ag0 : ag1;     // cross-modal agents

    // kv projection (WMMA), N = 512
    cvt(P[m][1], wkv, (size_t)2 * C_ * C_);
    k_gemm<<<dim3(MN / 32, (2 * C_) / 256), dim3(256), 0, stream>>>(
        src, wkv, nullptr, nullptr, kvb, 2 * C_, C_);

    // positional biases
    k_bias1<<<dim3((H_ * A_ * N_ + 255) / 256), dim3(256), 0, stream>>>(P[m][6], P[m][8], P[m][9], b1);
    k_bias2<<<dim3((H_ * N_ * A_ + 255) / 256), dim3(256), 0, stream>>>(P[m][7], P[m][10], P[m][11], b2);

    // stage 1: one wave per (b,h,a) -> agent_v
    k_stage1<<<dim3((B_ * H_ * A_ * 32 + 255) / 256), dim3(256), 0, stream>>>(agOther, kvb, b1, agv);

    // stage 2: queries -> branch output
    k_stage2<<<dim3((N_ + 255) / 256, H_, B_), dim3(256), 0, stream>>>(qq, agOther, agv, b2, ob);

    // depthwise conv add
    k_dwc<<<dim3((unsigned)((MNC + 255) / 256)), dim3(256), 0, stream>>>(kvb, P[m][4], P[m][5], ob);

    // output projection (WMMA) with fused bias, f32 straight to d_out
    cvt(ob, xb, MNC);                        // xb free after kv projections; reuse
    cvt(P[m][2], wpr, (size_t)C_ * C_);
    k_gemm<<<dim3(MN / 32, C_ / 256), dim3(256), 0, stream>>>(
        xb, wpr, P[m][3], out + (size_t)m * MNC, nullptr, C_, C_);
  }
}
